// ContextualAttention_31215822307689
// MI455X (gfx1250) — matrix operand order repaired and checked by reading, not compiled, not run on hardware
//
#include <hip/hip_runtime.h>

// ---------------------------------------------------------------------------
// Contextual attention for MI455X (gfx1250, wave32, WMMA bf16).
//   B=2, C=64, H=W=128, RATE=2 -> downsampled 64x64 (L=4096 positions),
//   K1 = C*3*3 = 576, patch dim NP = C*4*4 = 1024, SCALE=10.
// GEMMs use v_wmma_f32_16x16x32_bf16 with double-buffered LDS tiles filled by
// gfx1250 async DMA (global_load_async_to_lds_b128, ASYNCcnt-tracked).
// ---------------------------------------------------------------------------

typedef __attribute__((ext_vector_type(16))) __bf16 v16bf;
typedef __attribute__((ext_vector_type(8)))  float  v8f;

#define CDIM 64
#define HS   64
#define LDIM 4096      // HS*HS
#define K1   576       // CDIM*9
#define NP   1024      // CDIM*16
#define HF   128

__device__ __forceinline__ unsigned short f2bf(float f) {
  unsigned int x = __float_as_uint(f);
  x += 0x7FFFu + ((x >> 16) & 1u);          // round-to-nearest-even
  return (unsigned short)(x >> 16);
}

// Async memory->LDS 16B copy (gfx1250). LDS byte address = low 32 bits of the
// flat shared-aperture pointer.
__device__ __forceinline__ void async_copy16(void* lds_dst, const void* gsrc) {
  unsigned dst = (unsigned)(uintptr_t)lds_dst;
  asm volatile("global_load_async_to_lds_b128 %0, %1, off"
               :: "v"(dst), "v"(gsrc) : "memory");
}
__device__ __forceinline__ void wait_async0() {
  asm volatile("s_wait_asynccnt 0x0" ::: "memory");
}

// --------------------------- operand preparation ---------------------------

// A matrix: im2col of downsampled foreground, rows = pixel p, cols = (c,ki,kj)
__global__ void prep_A(const float* __restrict__ fg, unsigned short* __restrict__ Abf) {
  int t = blockIdx.x * blockDim.x + threadIdx.x;
  if (t >= LDIM * K1) return;
  int p = t / K1, q = t - p * K1;
  int c = q / 9, r = q - c * 9;
  int ki = r / 3, kj = r - ki * 3;
  int y = (p >> 6) + ki - 1, x = (p & 63) + kj - 1;
  float v = 0.f;
  if (y >= 0 && y < HS && x >= 0 && x < HS)
    v = fg[((size_t)c * HF + 2 * y) * HF + 2 * x];
  Abf[t] = f2bf(v);
}

// B matrix: L2-normalized 3x3 patches of downsampled background. One wave32
// per patch: 18 elements/lane, wave reduce sum-of-squares, normalize, emit bf16.
__global__ void prep_B(const float* __restrict__ bg, unsigned short* __restrict__ Bbf) {
  int l = blockIdx.x;
  int lane = threadIdx.x;               // 0..31
  int py = l >> 6, px = l & 63;
  float vals[18];
  float ss = 0.f;
#pragma unroll
  for (int t = 0; t < 18; ++t) {
    int k = lane + 32 * t;
    int c = k / 9, r = k - c * 9;
    int ki = r / 3, kj = r - ki * 3;
    int y = py + ki - 1, x = px + kj - 1;
    float v = 0.f;
    if (y >= 0 && y < HS && x >= 0 && x < HS)
      v = bg[((size_t)c * HF + 2 * y) * HF + 2 * x];
    vals[t] = v;
    ss += v * v;
  }
#pragma unroll
  for (int m = 16; m >= 1; m >>= 1) ss += __shfl_xor(ss, m, 32);
  float inv = 1.f / fmaxf(sqrtf(ss), 0.001f);
#pragma unroll
  for (int t = 0; t < 18; ++t)
    Bbf[(size_t)l * K1 + lane + 32 * t] = f2bf(vals[t] * inv);
}

// colsT[n][l]: transposed full-res 4x4 background patches (GEMM2 B operand)
__global__ void prep_colsT(const float* __restrict__ bg, unsigned short* __restrict__ colsT) {
  int t = blockIdx.x * blockDim.x + threadIdx.x;
  if (t >= NP * LDIM) return;
  int n = t / LDIM, l = t - n * LDIM;
  int c = n >> 4, a = (n >> 2) & 3, bb = n & 3;
  int Y = 2 * (l >> 6) + a - 1, X = 2 * (l & 63) + bb - 1;
  float v = 0.f;
  if (Y >= 0 && Y < HF && X >= 0 && X < HF)
    v = bg[((size_t)c * HF + Y) * HF + X];
  colsT[t] = f2bf(v);
}

// mask flag per patch l: 1 if 3x3 downsampled-mask patch sums to zero
__global__ void prep_mask(const float* __restrict__ mk, int* __restrict__ flag) {
  int l = blockIdx.x * blockDim.x + threadIdx.x;
  if (l >= LDIM) return;
  int py = l >> 6, px = l & 63;
  float s = 0.f;
#pragma unroll
  for (int ki = 0; ki < 3; ++ki)
#pragma unroll
    for (int kj = 0; kj < 3; ++kj) {
      int y = py + ki - 1, x = px + kj - 1;
      if (y >= 0 && y < HS && x >= 0 && x < HS)
        s += mk[(size_t)(2 * y) * HF + 2 * x];
    }
  flag[l] = (s == 0.f) ? 1 : 0;
}

// ------------------------------- WMMA GEMM ---------------------------------
// C[M][N] = A[M][K] * B[N][K]^T   (bf16 in, fp32 out)
// 128x128 block tile, BK=32, 8 wave32 waves, wave tile 64x32 = 8 WMMA accs.
// Double-buffered LDS tiles streamed with global_load_async_to_lds_b128.

#define BM 128
#define BN 128
#define BK 32
#define LDT 40      // LDS row stride in bf16 elems (80B: 16B aligned, bank-safe)

union FragU { uint4 u[2]; v16bf v; };

__launch_bounds__(256)
__global__ void gemm_bf16_nt(const unsigned short* __restrict__ A,
                             const unsigned short* __restrict__ B,
                             float* __restrict__ C,
                             int M, int N, int K) {
  __shared__ unsigned short As[2][BM * LDT];
  __shared__ unsigned short Bs[2][BN * LDT];

  const int tid  = threadIdx.x;
  const int wave = tid >> 5;
  const int lane = tid & 31;
  const int bm = blockIdx.y * BM;
  const int bn = blockIdx.x * BN;
  const int wm = (wave >> 2) * 64;      // wave grid 2 x 4
  const int wn = (wave & 3) * 32;

  const v8f vzero = {0.f, 0.f, 0.f, 0.f, 0.f, 0.f, 0.f, 0.f};
  v8f acc[4][2];
#pragma unroll
  for (int mi = 0; mi < 4; ++mi)
#pragma unroll
    for (int ni = 0; ni < 2; ++ni) acc[mi][ni] = vzero;

  // cooperative tile fill: each thread DMAs 32B/row-half of A and of B
  const int lr = tid >> 1;
  const int lh = (tid & 1) * 16;
  // fragment addressing per ISA VGPR layouts (wave32)
  const int laneM = lane & 15;
  const int kbA = (lane >> 4) * 8;      // A: K in {kbA..kbA+7, kbA+16..kbA+23}
  const int kbB = (lane >> 4) * 8;      // B: K in {kbB..kbB+7, kbB+16..kbB+23}

  const unsigned short* gA = A + (size_t)(bm + lr) * K + lh;
  const unsigned short* gB = B + (size_t)(bn + lr) * K + lh;
  unsigned short* sA0 = &As[0][lr * LDT + lh];
  unsigned short* sA1 = &As[1][lr * LDT + lh];
  unsigned short* sB0 = &Bs[0][lr * LDT + lh];
  unsigned short* sB1 = &Bs[1][lr * LDT + lh];

  // prologue: stream tile 0 into buffer 0
  async_copy16(sA0,     gA);
  async_copy16(sA0 + 8, gA + 8);
  async_copy16(sB0,     gB);
  async_copy16(sB0 + 8, gB + 8);
  wait_async0();
  __syncthreads();

  int buf = 0;
  for (int k0 = 0; k0 < K; k0 += BK) {
    const int nbuf = buf ^ 1;
    if (k0 + BK < K) {                  // stream next tile while computing
      unsigned short* dA = nbuf ? sA1 : sA0;
      unsigned short* dB = nbuf ? sB1 : sB0;
      async_copy16(dA,     gA + k0 + BK);
      async_copy16(dA + 8, gA + k0 + BK + 8);
      async_copy16(dB,     gB + k0 + BK);
      async_copy16(dB + 8, gB + k0 + BK + 8);
      if (k0 + 2 * BK < K) {            // gfx1250 global_prefetch two slabs out
        __builtin_prefetch(gA + k0 + 2 * BK, 0, 1);
        __builtin_prefetch(gB + k0 + 2 * BK, 0, 1);
      }
    }

    FragU af[4], bf[2];
#pragma unroll
    for (int mi = 0; mi < 4; ++mi) {
      const unsigned short* base = &As[buf][(wm + mi * 16 + laneM) * LDT + kbA];
      af[mi].u[0] = *reinterpret_cast<const uint4*>(base);        // K kbA..+7
      af[mi].u[1] = *reinterpret_cast<const uint4*>(base + 16);   // K kbA+16..+23
    }
#pragma unroll
    for (int ni = 0; ni < 2; ++ni) {
      const unsigned short* base = &Bs[buf][(wn + ni * 16 + laneM) * LDT + kbB];
      bf[ni].u[0] = *reinterpret_cast<const uint4*>(base);        // K kbB..+7
      bf[ni].u[1] = *reinterpret_cast<const uint4*>(base + 16);   // K kbB+16..+23
    }
#pragma unroll
    for (int mi = 0; mi < 4; ++mi)
#pragma unroll
      for (int ni = 0; ni < 2; ++ni)
        acc[mi][ni] = __builtin_amdgcn_wmma_f32_16x16x32_bf16(
            false, af[mi].v, false, bf[ni].v, (short)0, acc[mi][ni], false, false);

    wait_async0();                      // next tile resident in LDS
    __syncthreads();                    // all waves done reading `buf`
    buf = nbuf;
  }

  // C/D layout: elem e -> row = e + 8*(lane>>4), col = lane&15
#pragma unroll
  for (int mi = 0; mi < 4; ++mi)
#pragma unroll
    for (int ni = 0; ni < 2; ++ni) {
      int r0 = bm + wm + mi * 16 + 8 * (lane >> 4);
      int c0 = bn + wn + ni * 16 + laneM;
#pragma unroll
      for (int e = 0; e < 8; ++e)
        C[(size_t)(r0 + e) * N + c0] = acc[mi][ni][e];
    }
}

// --------------------- fused double diagonal propagation -------------------
// out = diagT( diag(in) ): 9-point gather with exact flat / transposed-flat
// boundary semantics of the reference.
__global__ void fuse_diag(const float* __restrict__ S, float* __restrict__ D) {
  int j = blockIdx.x * blockDim.x + threadIdx.x;
  int i = blockIdx.y;
  int ti = (i & 63) * 64 + (i >> 6);
  int tj = (j & 63) * 64 + (j >> 6);
  float s = 0.f;
#pragma unroll
  for (int d2 = -1; d2 <= 1; ++d2) {
    int a = ti + d2, b = tj + d2;
    if (a < 0 || a >= LDIM || b < 0 || b >= LDIM) continue;
    int i2 = (a & 63) * 64 + (a >> 6);
    int j2 = (b & 63) * 64 + (b >> 6);
#pragma unroll
    for (int d1 = -1; d1 <= 1; ++d1) {
      int ii = i2 + d1, jj = j2 + d1;
      if (ii < 0 || ii >= LDIM || jj < 0 || jj >= LDIM) continue;
      s += S[(size_t)ii * LDIM + jj];
    }
  }
  D[(size_t)i * LDIM + j] = s;
}

// ---------------------- mask + scaled softmax over patches -----------------
__global__ void softmax_rows(const float* __restrict__ S, const int* __restrict__ flag,
                             unsigned short* __restrict__ softBf) {
  __shared__ float red[256];
  int p = blockIdx.x;
  int tid = threadIdx.x;
  float v[16];
  float mx = -3.4e38f;
#pragma unroll
  for (int t = 0; t < 16; ++t) {
    int j = tid + 256 * t;
    float s = S[(size_t)p * LDIM + j];
    if (flag[j]) s = -1000.f;
    s *= 10.f;                           // SCALE
    v[t] = s;
    mx = fmaxf(mx, s);
  }
  red[tid] = mx; __syncthreads();
  for (int o = 128; o >= 1; o >>= 1) {
    if (tid < o) red[tid] = fmaxf(red[tid], red[tid + o]);
    __syncthreads();
  }
  mx = red[0]; __syncthreads();
  float sum = 0.f;
#pragma unroll
  for (int t = 0; t < 16; ++t) { v[t] = __expf(v[t] - mx); sum += v[t]; }
  red[tid] = sum; __syncthreads();
  for (int o = 128; o >= 1; o >>= 1) {
    if (tid < o) red[tid] += red[tid + o];
    __syncthreads();
  }
  float inv = 1.f / red[0];
#pragma unroll
  for (int t = 0; t < 16; ++t)
    softBf[(size_t)p * LDIM + tid + 256 * t] = f2bf(v[t] * inv);
}

// -------------------- transposed-conv overlap-add (gather) -----------------
// out[c,Y,X] = sum over (a,b) with Y=2y+a-1, X=2x+b-1 of P[p(y,x), c*16+a*4+b]
__global__ void gather_out(const float* __restrict__ P, float* __restrict__ out) {
  int t = blockIdx.x * blockDim.x + threadIdx.x;
  if (t >= CDIM * HF * HF) return;
  int c = t / (HF * HF);
  int rem = t - c * HF * HF;
  int Y = rem / HF, X = rem - Y * HF;
  float s = 0.f;
  int a0 = (Y + 1) & 1, b0 = (X + 1) & 1;
#pragma unroll
  for (int da = 0; da < 2; ++da) {
    int a = a0 + 2 * da;
    int yy = Y + 1 - a;
    if (yy < 0) continue;
    int y = yy >> 1;
    if (y >= HS) continue;
#pragma unroll
    for (int db = 0; db < 2; ++db) {
      int bb = b0 + 2 * db;
      int xx = X + 1 - bb;
      if (xx < 0) continue;
      int x = xx >> 1;
      if (x >= HS) continue;
      s += P[(size_t)(y * 64 + x) * NP + c * 16 + a * 4 + bb];
    }
  }
  out[t] = s;
}

// --------------------------------- driver ----------------------------------

extern "C" void kernel_launch(void* const* d_in, const int* in_sizes, int n_in,
                              void* d_out, int out_size, void* d_ws, size_t ws_size,
                              hipStream_t stream) {
  const float* FG = (const float*)d_in[0];
  const float* BG = (const float*)d_in[1];
  const float* MK = (const float*)d_in[2];
  float* OUT = (float*)d_out;

  const size_t MB = 1ull << 20;
  char* ws = (char*)d_ws;
  float*          S0     = (float*)(ws);                         // 64 MB scores
  float*          S1     = (float*)(ws + 64 * MB);               // 64 MB fused
  unsigned short* colsT  = (unsigned short*)(ws + 128 * MB);     // 8 MB
  unsigned short* Abf    = (unsigned short*)(ws + 136 * MB);     // 4.5 MB
  unsigned short* Bbf    = (unsigned short*)(ws + 141 * MB);     // 4.5 MB
  int*            mflag  = (int*)(ws + 146 * MB);                // 16 KB
  unsigned short* softBf = (unsigned short*)S0;                  // reuse (32 MB)
  float*          P      = S1;                                   // reuse (16 MB)

  if (ws_size < 147 * MB) return;

  for (int b = 0; b < 2; ++b) {
    const float* fg = FG + (size_t)b * CDIM * HF * HF;
    const float* bg = BG + (size_t)b * CDIM * HF * HF;
    const float* mk = MK + (size_t)b * HF * HF;
    float* outb = OUT + (size_t)b * CDIM * HF * HF;

    prep_A    <<<(LDIM * K1 + 255) / 256, 256, 0, stream>>>(fg, Abf);
    prep_B    <<<LDIM, 32, 0, stream>>>(bg, Bbf);
    prep_colsT<<<(NP * LDIM + 255) / 256, 256, 0, stream>>>(bg, colsT);
    prep_mask <<<(LDIM + 255) / 256, 256, 0, stream>>>(mk, mflag);

    // scores[p, l] = <fg-neigh(p), bg-patch(l)>   (4096 x 4096 x 576)
    gemm_bf16_nt<<<dim3(LDIM / BN, LDIM / BM), 256, 0, stream>>>(
        Abf, Bbf, S0, LDIM, LDIM, K1);

    fuse_diag   <<<dim3(LDIM / 256, LDIM), 256, 0, stream>>>(S0, S1);
    softmax_rows<<<LDIM, 256, 0, stream>>>(S1, mflag, softBf);

    // P[p, n] = sum_l soft[p, l] * colsT[n, l]    (4096 x 1024 x 4096)
    gemm_bf16_nt<<<dim3(NP / BN, LDIM / BM), 256, 0, stream>>>(
        softBf, colsT, P, LDIM, NP, LDIM);

    gather_out<<<(CDIM * HF * HF + 255) / 256, 256, 0, stream>>>(P, outb);
  }
}
